// VectorQuantizer_48576080118546
// MI455X (gfx1250) — matrix operand order repaired and checked by reading, not compiled, not run on hardware
//
#include <hip/hip_runtime.h>

#define NUM_CODES 4096
#define DIM 256
#define M_ROWS 32768   // 8 * 4096
#define BETA 0.25f
#define LDS_PITCH (DIM + 8)   // 264 ushorts = 528 B per row -> 16B aligned, conflict-free

typedef __attribute__((ext_vector_type(16))) __bf16 v16bf;
typedef __attribute__((ext_vector_type(8)))  float  v8f;

union Frag16 {
  v16bf v;
  unsigned short s[16];
  uint4 q[2];
};

__device__ __forceinline__ unsigned short f2bf(float f) {
  unsigned int u = __float_as_uint(f);
  u += 0x7FFFu + ((u >> 16) & 1u);          // round-to-nearest-even
  return (unsigned short)(u >> 16);
}
__device__ __forceinline__ float bf2f(unsigned short h) {
  return __uint_as_float(((unsigned int)h) << 16);
}

// Truncate a flat shared-memory pointer to the 32-bit LDS byte address
__device__ __forceinline__ unsigned lds_off(const void* p) {
  return (unsigned)(size_t)p;
}

// Issue one lane's share of a 16-code tile (64 B of hi + 64 B of lo) as
// async global->LDS DMA. offset: is added to BOTH the LDS and global address.
__device__ __forceinline__ void issue_tile_async(const unsigned short* gh,
                                                 const unsigned short* gl,
                                                 unsigned dh, unsigned dl) {
  asm volatile("global_load_async_to_lds_b128 %0, %1, off"            :: "v"(dh), "v"(gh) : "memory");
  asm volatile("global_load_async_to_lds_b128 %0, %1, off offset:16"  :: "v"(dh), "v"(gh) : "memory");
  asm volatile("global_load_async_to_lds_b128 %0, %1, off offset:32"  :: "v"(dh), "v"(gh) : "memory");
  asm volatile("global_load_async_to_lds_b128 %0, %1, off offset:48"  :: "v"(dh), "v"(gh) : "memory");
  asm volatile("global_load_async_to_lds_b128 %0, %1, off"            :: "v"(dl), "v"(gl) : "memory");
  asm volatile("global_load_async_to_lds_b128 %0, %1, off offset:16"  :: "v"(dl), "v"(gl) : "memory");
  asm volatile("global_load_async_to_lds_b128 %0, %1, off offset:32"  :: "v"(dl), "v"(gl) : "memory");
  asm volatile("global_load_async_to_lds_b128 %0, %1, off offset:48"  :: "v"(dl), "v"(gl) : "memory");
}

__device__ __forceinline__ void wait_async_zero() {
  asm volatile("s_wait_asynccnt 0x0" ::: "memory");
}

// ---------------------------------------------------------------------------
// Phase 1: codebook f32 -> bf16 hi/lo split + ||c_k||^2
// ---------------------------------------------------------------------------
__global__ void vq_prep_codebook(const float* __restrict__ cb,
                                 unsigned short* __restrict__ chi,
                                 unsigned short* __restrict__ clo,
                                 float* __restrict__ cnorm) {
  const int k = blockIdx.x;
  const int d = threadIdx.x;
  const float v = cb[(size_t)k * DIM + d];
  const unsigned short h = f2bf(v);
  chi[(size_t)k * DIM + d] = h;
  clo[(size_t)k * DIM + d] = f2bf(v - bf2f(h));

  float s = v * v;
  for (int m = 16; m >= 1; m >>= 1) s += __shfl_xor(s, m, 32);
  __shared__ float red[8];
  if ((d & 31) == 0) red[d >> 5] = s;
  __syncthreads();
  if (d == 0) {
    float t = 0.f;
    for (int i = 0; i < 8; ++i) t += red[i];
    cnorm[k] = t;
  }
}

// ---------------------------------------------------------------------------
// Phase 2: WMMA distance + argmin. One wave owns a 16-row M-tile; 4 waves per
// block share double-buffered, async-DMA'd 16-code B tiles in LDS.
// 3 WMMAs per K-chunk (bf16 hi/lo split => ~1e-6 accuracy on distances).
// ---------------------------------------------------------------------------
__global__ __launch_bounds__(128, 1)
void vq_argmin(const float* __restrict__ x,
               const unsigned short* __restrict__ chi,
               const unsigned short* __restrict__ clo,
               const float* __restrict__ cnorm,
               int* __restrict__ out_idx) {
  __shared__ unsigned short lds_hi[2][16 * LDS_PITCH];
  __shared__ unsigned short lds_lo[2][16 * LDS_PITCH];

  const int tid  = threadIdx.x;
  const int wave = tid >> 5;
  const int lane = tid & 31;
  const int hgrp = lane >> 4;   // half-group 0/1
  const int l15  = lane & 15;

  const int mtile = blockIdx.x * 4 + wave;
  const int mbase = mtile * 16;

  // Load this wave's A tile (16 x 256 f32) and split into bf16 hi/lo fragments.
  // A layout (16-bit, 16x32): elem j <- K = 16*hgrp + j
  Frag16 axh[8], axl[8];
  {
    const float* xrow = x + (size_t)(mbase + l15) * DIM;
    for (int c = 0; c < 8; ++c) {
      const int d0 = 32 * c + 16 * hgrp;
      const float4* p0 = reinterpret_cast<const float4*>(xrow + d0);
      const float4* p1 = reinterpret_cast<const float4*>(xrow + d0 + 8);
      float4 a0 = p0[0], a1 = p0[1];
      float4 b0 = p1[0], b1 = p1[1];
      const float f[16] = {a0.x, a0.y, a0.z, a0.w, a1.x, a1.y, a1.z, a1.w,
                           b0.x, b0.y, b0.z, b0.w, b1.x, b1.y, b1.z, b1.w};
      for (int j = 0; j < 16; ++j) {
        const unsigned short h = f2bf(f[j]);
        axh[c].s[j] = h;
        axl[c].s[j] = f2bf(f[j] - bf2f(h));
      }
    }
  }

  float best_v[8];
  int   best_i[8];
  for (int v = 0; v < 8; ++v) { best_v[v] = 3.0e38f; best_i[v] = 0; }

  // Per-thread staging share: 32 ushorts (64 B) of hi and of lo per tile.
  const int srow = tid >> 3;         // code row within tile (0..15)
  const int sseg = (tid & 7) * 32;   // ushort column (0..224)
  const unsigned selem = (unsigned)(srow * LDS_PITCH + sseg) * 2u;
  const unsigned dh0 = lds_off(&lds_hi[0][0]) + selem;
  const unsigned dh1 = lds_off(&lds_hi[1][0]) + selem;
  const unsigned dl0 = lds_off(&lds_lo[0][0]) + selem;
  const unsigned dl1 = lds_off(&lds_lo[1][0]) + selem;

  // Prologue: kick off DMA for tile 0 into buffer 0.
  issue_tile_async(chi + (size_t)srow * DIM + sseg,
                   clo + (size_t)srow * DIM + sseg, dh0, dl0);

  for (int nt = 0; nt < NUM_CODES / 16; ++nt) {
    const int nbase = nt * 16;
    const int buf = nt & 1;

    wait_async_zero();     // this wave's DMA for the current buffer has landed
    __syncthreads();       // all waves' DMAs landed; other buffer free to overwrite

    if (nt + 1 < NUM_CODES / 16) {
      const size_t g = (size_t)(nbase + 16 + srow) * DIM + sseg;
      issue_tile_async(chi + g, clo + g,
                       buf ? dh0 : dh1, buf ? dl0 : dl1);
    }

    const unsigned short* bhi = &lds_hi[buf][0];
    const unsigned short* blo = &lds_lo[buf][0];

    v8f acc = {0.f, 0.f, 0.f, 0.f, 0.f, 0.f, 0.f, 0.f};
    for (int c = 0; c < 8; ++c) {
      // B layout (32x16): lane -> N = l15, elem i <- K = 16*hgrp + i (contig 32B)
      Frag16 bh, bl;
      const uint4* ph = reinterpret_cast<const uint4*>(bhi + l15 * LDS_PITCH + 32 * c + 16 * hgrp);
      const uint4* pl = reinterpret_cast<const uint4*>(blo + l15 * LDS_PITCH + 32 * c + 16 * hgrp);
      bh.q[0] = ph[0]; bh.q[1] = ph[1];
      bl.q[0] = pl[0]; bl.q[1] = pl[1];
      acc = __builtin_amdgcn_wmma_f32_16x16x32_bf16(false, axh[c].v, false, bh.v, (short)0, acc, false, false);
      acc = __builtin_amdgcn_wmma_f32_16x16x32_bf16(false, axh[c].v, false, bl.v, (short)0, acc, false, false);
      acc = __builtin_amdgcn_wmma_f32_16x16x32_bf16(false, axl[c].v, false, bh.v, (short)0, acc, false, false);
    }

    const float cn = cnorm[nbase + l15];
    const int   ci = nbase + l15;
    for (int v = 0; v < 8; ++v) {
      const float dv = cn - 2.0f * acc[v];   // ||x||^2 is row-constant: dropped
      if (dv < best_v[v]) { best_v[v] = dv; best_i[v] = ci; }
    }
  }

  // Reduce min/argmin across the 16 lanes of this half-group (C: M = v + 8*hgrp).
  for (int mask = 8; mask >= 1; mask >>= 1) {
    for (int v = 0; v < 8; ++v) {
      const float ov = __shfl_xor(best_v[v], mask, 32);
      const int   oi = __shfl_xor(best_i[v], mask, 32);
      if (ov < best_v[v] || (ov == best_v[v] && oi < best_i[v])) {
        best_v[v] = ov; best_i[v] = oi;
      }
    }
  }
  if (l15 == 0) {
    for (int v = 0; v < 8; ++v)
      out_idx[mbase + v + 8 * hgrp] = best_i[v];
  }
}

// ---------------------------------------------------------------------------
// Phase 3: gather z_q, straight-through output, per-row loss partial, indices.
// ---------------------------------------------------------------------------
__global__ void vq_gather(const float* __restrict__ x,
                          const float* __restrict__ cb,
                          const int* __restrict__ idx,
                          float* __restrict__ zq,
                          float* __restrict__ idx_f,
                          float* __restrict__ partial) {
  const int row = blockIdx.x;
  const int d   = threadIdx.x;
  const int k   = idx[row];
  const float xv = x[(size_t)row * DIM + d];
  const float cv = cb[(size_t)k * DIM + d];
  zq[(size_t)row * DIM + d] = xv + (cv - xv);   // x + sg(z_q - x), as reference
  const float diff = cv - xv;

  float s = diff * diff;
  for (int m = 16; m >= 1; m >>= 1) s += __shfl_xor(s, m, 32);
  __shared__ float red[8];
  if ((d & 31) == 0) red[d >> 5] = s;
  __syncthreads();
  if (d == 0) {
    float t = 0.f;
    for (int i = 0; i < 8; ++i) t += red[i];
    partial[row] = t;
    idx_f[row] = (float)k;
  }
}

__global__ void vq_finalize(const float* __restrict__ partial,
                            float* __restrict__ loss) {
  const int t = threadIdx.x;
  float s = 0.f;
  for (int i = t; i < M_ROWS; i += 256) s += partial[i];
  for (int m = 16; m >= 1; m >>= 1) s += __shfl_xor(s, m, 32);
  __shared__ float red[8];
  if ((t & 31) == 0) red[t >> 5] = s;
  __syncthreads();
  if (t == 0) {
    float tot = 0.f;
    for (int i = 0; i < 8; ++i) tot += red[i];
    loss[0] = (1.0f + BETA) * tot / (float)(M_ROWS * DIM);
  }
}

// ---------------------------------------------------------------------------
extern "C" void kernel_launch(void* const* d_in, const int* in_sizes, int n_in,
                              void* d_out, int out_size, void* d_ws, size_t ws_size,
                              hipStream_t stream) {
  const float* x  = (const float*)d_in[0];   // [8, 4096, 256] f32
  const float* cb = (const float*)d_in[1];   // [4096, 256]    f32

  char* ws = (char*)d_ws;
  unsigned short* chi   = (unsigned short*)(ws);              // 2 MB
  unsigned short* clo   = (unsigned short*)(ws + 2097152);    // 2 MB
  float*          cnorm = (float*)(ws + 4194304);             // 16 KB
  int*            idx   = (int*)(ws + 4210688);               // 128 KB
  float*          part  = (float*)(ws + 4341760);             // 128 KB

  float* out  = (float*)d_out;
  float* zq   = out;                               // 8*4096*256
  float* loss = out + (size_t)M_ROWS * DIM;        // 1
  float* idxf = loss + 1;                          // 8*4096 (indices as float)

  vq_prep_codebook<<<NUM_CODES, 256, 0, stream>>>(cb, chi, clo, cnorm);
  vq_argmin<<<(M_ROWS / 16) / 4, 128, 0, stream>>>(x, chi, clo, cnorm, idx);
  vq_gather<<<M_ROWS, DIM, 0, stream>>>(x, cb, idx, zq, idxf, part);
  vq_finalize<<<1, 256, 0, stream>>>(part, loss);
}
